// LateralLayerEfficientNetwork1L_7026566496887
// MI455X (gfx1250) — compile-verified, matrix operand order checked by reading, not run on hardware
//
#include <hip/hip_runtime.h>

// ---------------------------------------------------------------------------
// LateralLayerEfficientNetwork1L on MI455X (gfx1250, wave32, WMMA + TDM)
//
// x: [8,128,64,64] f32, W: [128,256,11,11] f32
// out: changes[6] f32 ++ features[8,6,128,64,64] f32
//
//   K0 init_state   : zero bias / counts / change (ws)
//   K1 pack_weights : W -> bf16 pre-swizzled WMMA B-fragments
//                     [tap 121][ciChunk 8][coTile 8][lane 32][16]
//   K2 pack_x       : x -> bf16 transposed rows  xb[b][y][xs][ci]
//   K3 conv_row(-1) : Y0 = conv(x, W[:,:128]) via WMMA; A rows staged by TDM
//   per t in 0..5:
//     K3 conv_row(t): Yz = conv(z_{t-1}, W[:,128:]) (TDM-staged from zb),
//                     fused: +Y0 -bias, EMA, channel-max k-WTA, z->features,
//                     z->zb (bf16, transposed) for next step, rate/change stats
//     K4 update_stats
// ---------------------------------------------------------------------------

typedef __attribute__((ext_vector_type(16))) __bf16 v16bf;
typedef __attribute__((ext_vector_type(8)))  float  v8f;
typedef __attribute__((ext_vector_type(4)))  unsigned int u32x4;
typedef __attribute__((ext_vector_type(8)))  int i32x8;
typedef __attribute__((ext_vector_type(4)))  int i32x4;

union FragU {
  uint4  q[2];   // 32 bytes
  v16bf  v;
};

#define HW        4096
#define NCH       128
#define NB        8
#define NT        6
#define ROWE      8192                                      // 64 px * 128 ci
#define WPK_BYTES ((size_t)121 * 8 * 8 * 32 * 16 * 2)       // 7.93 MB
#define Y0_BYTES  ((size_t)NB * NCH * HW * 4)               // 16.78 MB
#define XB_BYTES  ((size_t)NB * 64 * ROWE * 2)              // 8.39 MB
#define ZB_BYTES  ((size_t)NB * 64 * ROWE * 2)              // 8.39 MB

__device__ __forceinline__ unsigned short f2bf(float f) {
  unsigned int u = __builtin_bit_cast(unsigned int, f);
  unsigned int r = u + 0x7FFFu + ((u >> 16) & 1u);          // round-nearest-even
  return (unsigned short)(r >> 16);
}

// TDM: copy one transposed bf16 row (8192 elts, 16 KB) global -> LDS,
// inserting 16 B of padding after every 256 B so the LDS row pitch is 272 B.
__device__ __forceinline__ void tdm_row_to_lds(unsigned lds_off,
                                               unsigned long long ga) {
  u32x4 g0;
  g0[0] = 1u;                                         // count=1, no gather
  g0[1] = lds_off;                                    // lds_addr (bytes)
  g0[2] = (unsigned)(ga & 0xFFFFFFFFull);             // global_addr[31:0]
  g0[3] = (unsigned)((ga >> 32) & 0x1FFFFFFull)       // global_addr[56:32]
          | (2u << 30);                               // type=2 ("image")
  i32x8 g1;
  g1[0] = (int)((1u << 16)      // data_size = 2 B
              | (1u << 20)      // pad_enable
              | (5u << 22)      // pad_interval: 64 DWORDs (256 B)
              | (3u << 25));    // pad_amount:   4 DWORDs (16 B)
  g1[1] = (int)(8192u << 16);   // tensor_dim0[15:0] in [31:16]
  g1[2] = (int)(1u << 16);      // tensor_dim0 hi = 0 | tensor_dim1 = 1
  g1[3] = (int)(8192u << 16);   // tensor_dim1 hi = 0 | tile_dim0 = 8192
  g1[4] = 1;                    // tile_dim1 = 1, tile_dim2 = 0
  g1[5] = 8192;                 // tensor_dim0_stride low32
  g1[6] = 0;
  g1[7] = 0;
  i32x4 z4 = {0, 0, 0, 0};
#if __clang_major__ >= 23
  i32x8 z8 = {0, 0, 0, 0, 0, 0, 0, 0};
  __builtin_amdgcn_tensor_load_to_lds(g0, g1, z4, z4, z8, 0);
#else
  __builtin_amdgcn_tensor_load_to_lds(g0, g1, z4, z4, 0);
#endif
}

// ---------------------------------------------------------------------------
__global__ __launch_bounds__(128) void init_state(float* bias, float* counts,
                                                  float* change) {
  int i = threadIdx.x;
  bias[i] = 0.0f;
  counts[i] = 0.0f;
  if (i == 0) change[0] = 0.0f;
}

// ---------------------------------------------------------------------------
// B-fragment packing: lane l holds column n = l&15, half h = l>>4,
// element e -> k = ((e<8)? e : e+8) + 8*h (consistent with the A layout).
__global__ __launch_bounds__(256) void pack_weights(const float* __restrict__ W,
                                                    unsigned short* __restrict__ Wpk) {
  const int tap = blockIdx.x;          // 0..120
  const int cc  = blockIdx.y;          // 0..7
  const int ct  = threadIdx.x >> 5;    // 0..7
  const int l   = threadIdx.x & 31;
  const int ky  = tap / 11, kx = tap % 11;

  alignas(16) unsigned short tmp[16];
#pragma unroll
  for (int e = 0; e < 16; ++e) {
    int k  = ((e < 8) ? e : (e + 8)) + ((l >> 4) << 3);
    int ci = cc * 32 + k;
    int co = ct * 16 + (l & 15);
    tmp[e] = f2bf(W[(((size_t)co * 256 + ci) * 11 + ky) * 11 + kx]);
  }
  uint4* dst = (uint4*)(Wpk + ((((size_t)tap * 8 + cc) * 8 + ct) * 32 + l) * 16);
  dst[0] = *(const uint4*)&tmp[0];
  dst[1] = *(const uint4*)&tmp[8];
}

// ---------------------------------------------------------------------------
// x [b][ci][y][xs] f32  ->  xb [b][y][xs][ci] bf16 (TDM-ready transposed rows)
__global__ __launch_bounds__(256) void pack_x(const float* __restrict__ x,
                                              unsigned short* __restrict__ xb) {
  const int y = blockIdx.x, b = blockIdx.y;
  unsigned short* dst = xb + (size_t)(b * 64 + y) * ROWE;
  const float* src = x + (size_t)b * NCH * HW + y * 64;
  for (int j = 0; j < 32; ++j) {
    int lin = threadIdx.x + (j << 8);
    int ci = lin & 127, xs = lin >> 7;
    dst[xs * 128 + ci] = f2bf(src[(size_t)ci * HW + xs]);
  }
}

// ---------------------------------------------------------------------------
// One block = one (batch b, output row y). 8 waves; wave w owns co-tile w.
// t == -1 : x-conv mode (input xb, ci chunk 0..3) -> writes Y0.
// t >=  0 : step t (input zb, ci chunk 4..7) -> fused epilogue.
__global__ __launch_bounds__(256) void conv_row(
    const unsigned short* __restrict__ xb, const unsigned short* __restrict__ zbin,
    unsigned short* __restrict__ zbout, const unsigned short* __restrict__ Wpk,
    float* __restrict__ Y0, const float* __restrict__ bias,
    float* __restrict__ dout, float* __restrict__ counts,
    float* __restrict__ change, int t) {
  // LDS: input row stage [74 cols][136 (128 ci + 8 pad)] bf16, pitch 272 B
  __shared__ unsigned short sStage[74 * 136];    // 20,128 B
  __shared__ float sY[NCH * 65];                 // [co][px] padded, 33,280 B
  __shared__ float sMax[64];
  __shared__ float sRed[128];

  const int y    = blockIdx.x;
  const int b    = blockIdx.y;
  const int tid  = threadIdx.x;
  const int wv   = tid >> 5;
  const int lane = tid & 31;
  const int ln   = lane & 15;
  const int lh   = lane >> 4;

  const bool xmode  = (t < 0);
  const bool doConv = xmode || (t > 0);
  const unsigned short* inB = xmode ? xb : zbin;
  const int ccBase = xmode ? 0 : 4;

  float* feat = dout + NT;

  // zero the 5-column left/right halo once (persists across ky)
  for (int lin = tid; lin < 10 * 128; lin += 256) {
    int p = lin >> 7;
    int col = (p < 5) ? p : (64 + p);
    sStage[col * 136 + (lin & 127)] = 0;
  }

  v8f acc[4] = {};

  if (doConv) {
    const unsigned lds_int = (unsigned)(size_t)&sStage[5 * 136];
    for (int ky = 0; ky < 11; ++ky) {
      const int yy = y + ky - 5;
      const bool rowok = (yy >= 0) && (yy < 64);
      __syncthreads();                       // prior-iteration readers done
      if (rowok && wv == 0) {
        tdm_row_to_lds(lds_int,
            (unsigned long long)(size_t)(inB + (size_t)(b * 64 + yy) * ROWE));
        __builtin_amdgcn_s_wait_tensorcnt(0);
      }
      __syncthreads();
      if (!rowok) continue;                  // zero-padding rows contribute 0
      for (int kx = 0; kx < 11; ++kx) {
        const int tap = ky * 11 + kx;
#pragma unroll
        for (int cc = 0; cc < 4; ++cc) {
          const uint4* bp = (const uint4*)(Wpk +
              ((((size_t)tap * 8 + (ccBase + cc)) * 8 + wv) * 32 + lane) * 16);
          FragU bu;
          bu.q[0] = bp[0];
          bu.q[1] = bp[1];
#pragma unroll
          for (int mt = 0; mt < 4; ++mt) {
            const int xcol = mt * 16 + ln + kx;          // [0,74)
            const unsigned short* as =
                &sStage[xcol * 136 + cc * 32 + (lh << 3)];
            FragU au;
            au.q[0] = *(const uint4*)as;
            au.q[1] = *(const uint4*)(as + 16);
            acc[mt] = __builtin_amdgcn_wmma_f32_16x16x32_bf16(
                false, au.v, false, bu.v, (short)0, acc[mt], false, false);
          }
        }
      }
    }
  }

  // deposit accumulators into sY[co][px] (D layout: m = r + 8*lh, n = ln)
  __syncthreads();
#pragma unroll
  for (int mt = 0; mt < 4; ++mt)
#pragma unroll
    for (int r = 0; r < 8; ++r)
      sY[(wv * 16 + ln) * 65 + (mt * 16 + r + (lh << 3))] = acc[mt][r];
  __syncthreads();

  if (xmode) {
    float* y0p = Y0 + (size_t)b * NCH * HW + y * 64;
    for (int j = 0; j < 32; ++j) {
      int lin = tid + (j << 8);
      int co = lin >> 6, px = lin & 63;
      y0p[(size_t)co * HW + px] = sY[co * 65 + px];
    }
    return;
  }

  // ---- step epilogue: realy = Y0 + Yz - bias, EMA with z_prev ----
  const float* y0p   = Y0 + (size_t)b * NCH * HW + y * 64;
  const float* zprev = feat + (size_t)(b * NT + (t > 0 ? t - 1 : 0)) * NCH * HW
                       + y * 64;
  for (int j = 0; j < 32; ++j) {
    int lin = tid + (j << 8);
    int co = lin >> 6, px = lin & 63;
    float v = sY[co * 65 + px] + y0p[(size_t)co * HW + px] - bias[co];
    if (t > 0) v = 0.5f * v + 0.5f * zprev[(size_t)co * HW + px];
    sY[co * 65 + px] = v;
  }
  __syncthreads();

  // per-pixel channel max (k-WTA, k=1)
  if (tid < 64) {
    float m = -3.4e38f;
    for (int co = 0; co < NCH; ++co) m = fmaxf(m, sY[co * 65 + tid]);
    sMax[tid] = m;
  }
  __syncthreads();

  // binarize: z -> features[:, t] (f32, coalesced along px)
  float* zout = feat + (size_t)(b * NT + t) * NCH * HW + y * 64;
  for (int j = 0; j < 32; ++j) {
    int lin = tid + (j << 8);
    int co = lin >> 6, px = lin & 63;
    float v = sY[co * 65 + px];
    float z = (v >= sMax[px] && v > 0.0f) ? 1.0f : 0.0f;
    zout[(size_t)co * HW + px] = z;
  }
  // z -> zb (bf16, transposed rows, TDM source for step t+1)
  unsigned short* zbp = zbout + (size_t)(b * 64 + y) * ROWE;
  for (int j = 0; j < 32; ++j) {
    int lin = tid + (j << 8);
    int ci = lin & 127, px = lin >> 7;
    float v = sY[ci * 65 + px];
    float z = (v >= sMax[px] && v > 0.0f) ? 1.0f : 0.0f;
    zbp[px * 128 + ci] = f2bf(z);
  }

  // per-channel rate counts + L1 change, block-reduced before atomics
  float chg = 0.0f;
  if (tid < NCH) {
    float cnt = 0.0f;
    for (int px = 0; px < 64; ++px) {
      float v = sY[tid * 65 + px];
      float z = (v >= sMax[px] && v > 0.0f) ? 1.0f : 0.0f;
      float zp = (t > 0) ? zprev[(size_t)tid * HW + px] : 0.0f;
      cnt += z;
      chg += fabsf(z - zp);
    }
    atomicAdd(&counts[tid], cnt);
    sRed[tid] = chg;
  }
  __syncthreads();
  for (int s = 64; s > 0; s >>= 1) {
    if (tid < s) sRed[tid] += sRed[tid + s];
    __syncthreads();
  }
  if (tid == 0) atomicAdd(change, sRed[0]);
}

// ---------------------------------------------------------------------------
__global__ __launch_bounds__(128) void update_stats(float* bias, float* counts,
                                                    float* change,
                                                    float* changes_out, int t) {
  int co = threadIdx.x;
  float rate = counts[co] / (float)(NB * HW);
  bias[co] += 0.05f * (rate - 1.0f / 128.0f);
  counts[co] = 0.0f;
  if (co == 0) {
    changes_out[t] = change[0] / (float)((size_t)NB * 256 * HW);
    change[0] = 0.0f;
  }
}

// ---------------------------------------------------------------------------
extern "C" void kernel_launch(void* const* d_in, const int* in_sizes, int n_in,
                              void* d_out, int out_size, void* d_ws,
                              size_t ws_size, hipStream_t stream) {
  const float* x = (const float*)d_in[0];
  const float* W = (const float*)d_in[1];
  float* out = (float*)d_out;

  char* ws = (char*)d_ws;
  unsigned short* Wpk = (unsigned short*)ws;
  float* Y0           = (float*)(ws + WPK_BYTES);
  unsigned short* xb  = (unsigned short*)(ws + WPK_BYTES + Y0_BYTES);
  unsigned short* zb  = (unsigned short*)(ws + WPK_BYTES + Y0_BYTES + XB_BYTES);
  float* bias   = (float*)(ws + WPK_BYTES + Y0_BYTES + XB_BYTES + ZB_BYTES);
  float* counts = bias + 128;
  float* change = counts + 128;

  hipLaunchKernelGGL(init_state, dim3(1), dim3(128), 0, stream,
                     bias, counts, change);
  hipLaunchKernelGGL(pack_weights, dim3(121, 8), dim3(256), 0, stream, W, Wpk);

  dim3 grid(64, 8);   // (row, batch)
  hipLaunchKernelGGL(pack_x, grid, dim3(256), 0, stream, x, xb);

  // one-time dense x-half conv -> Y0
  hipLaunchKernelGGL(conv_row, grid, dim3(256), 0, stream,
                     xb, zb, zb, Wpk, Y0, bias, out, counts, change, -1);

  for (int t = 0; t < NT; ++t) {
    hipLaunchKernelGGL(conv_row, grid, dim3(256), 0, stream,
                       xb, zb, zb, Wpk, Y0, bias, out, counts, change, t);
    hipLaunchKernelGGL(update_stats, dim3(1), dim3(128), 0, stream,
                       bias, counts, change, out, t);
  }
}
